// Net_81527069213046
// MI455X (gfx1250) — matrix operand order repaired and checked by reading, not compiled, not run on hardware
//
#include <hip/hip_runtime.h>
#include <hip/hip_bf16.h>

typedef __attribute__((ext_vector_type(16))) _Float16 v16h;
typedef __attribute__((ext_vector_type(8)))  float    v8f;

#define N_NODES 50000
#define N_EDGES 1600000

__device__ __forceinline__ float sigmoidf_(float x) {
    return 1.0f / (1.0f + __expf(-x));
}

// ---------- zero ----------
__global__ void zero_f32(float* p, int n) {
    int i = blockIdx.x * blockDim.x + threadIdx.x;
    if (i < n) p[i] = 0.0f;
}

// ---------- weight prep ----------
// WihAs: fragment-ready swizzled B for the big GEMM.
//   index = ((chunk*16 + nt)*32 + lane)*16 + t  ->  (f16) Wih[col*192 + k]
//   with k = chunk*32 + (lane>>4)*8 + (t&7) + (t>>3)*16, col = nt*16 + (lane&15)
// W2s:  fragment-ready swizzled B for SAGE2 (K=32, N=64):
//   index = (nt*32 + lane)*16 + t  ->  k = (lane>>4)*8 + (t&7) + (t>>3)*16, col = nt*16 + (lane&15)
// WihS/WihT: [s][r] transposed one-hot column tables (f32)
__global__ void prep_weights(const float* Wih, const float* c2_Wl, const float* c2_Wr,
                             _Float16* WihAs, float* WihS, float* WihT, _Float16* W2s) {
    int idx = blockIdx.x * blockDim.x + threadIdx.x;
    if (idx < 16384) {
        // big-GEMM B fragments
        int t = idx & 15;
        int lane = (idx >> 4) & 31;
        int nt = (idx >> 9) & 15;
        int chunk = (idx >> 13) & 1;
        int k = chunk * 32 + (lane >> 4) * 8 + (t & 7) + (t >> 3) * 16;
        int col = nt * 16 + (lane & 15);
        WihAs[idx] = (_Float16)Wih[col * 192 + k];
        // one-hot gather tables
        int kk = idx >> 8, r = idx & 255;
        WihS[kk * 256 + r] = Wih[r * 192 + 64 + kk];
        WihT[kk * 256 + r] = Wih[r * 192 + 128 + kk];
    }
    if (idx < 2048) {
        int t = idx & 15;
        int lane = (idx >> 4) & 31;
        int nt = (idx >> 9) & 3;
        int k = (lane >> 4) * 8 + (t & 7) + (t >> 3) * 16;
        int col = nt * 16 + (lane & 15);
        float v = (k < 16) ? c2_Wl[col * 16 + k] : c2_Wr[col * 16 + (k - 16)];
        W2s[idx] = (_Float16)v;
    }
}

// ---------- SAGE layer 1: edge scatter (sum + degree) ----------
__global__ void sage1_edges(const int* edge_index, const float* x, float* sum1, float* cnt) {
    int e = blockIdx.x * blockDim.x + threadIdx.x;
    if (e >= N_EDGES) return;
    int src = edge_index[e];
    int dst = edge_index[N_EDGES + e];
    atomicAdd(&sum1[dst], x[src]);
    atomicAdd(&cnt[dst], 1.0f);
}

// ---------- SAGE layer 1: node update ----------
__global__ void sage1_nodes(const float* x, const float* sum1, const float* cnt,
                            const float* Wl, const float* bl, const float* Wr, float* h1) {
    int idx = blockIdx.x * blockDim.x + threadIdx.x;
    if (idx >= N_NODES * 16) return;
    int i = idx >> 4, j = idx & 15;
    float mean = sum1[i] / fmaxf(cnt[i], 1.0f);
    float v = mean * Wl[j] + bl[j] + x[i] * Wr[j];
    h1[idx] = fmaxf(v, 0.0f);
}

// ---------- SAGE layer 2: edge scatter of 16-wide features ----------
__global__ void sage2_edges(const int* edge_index, const float* h1, float* sum2) {
    long long idx = (long long)blockIdx.x * blockDim.x + threadIdx.x;
    if (idx >= (long long)N_EDGES * 16) return;
    int e = (int)(idx >> 4), f = (int)(idx & 15);
    int src = edge_index[e];
    int dst = edge_index[N_EDGES + e];
    atomicAdd(&sum2[dst * 16 + f], h1[src * 16 + f]);
}

// ---------- SAGE layer 2 node update via WMMA:
//   h2 = relu([mean2 | h1](50000x32) @ W2(32x64) + c2_bl)
__global__ __launch_bounds__(256) void sage2_wmma(const float* sum2, const float* cnt,
                                                  const float* h1, const _Float16* W2s,
                                                  const float* c2_bl, float* h2) {
    int wave = (blockIdx.x * blockDim.x + threadIdx.x) >> 5;
    int lane = threadIdx.x & 31;
    if (wave >= N_NODES / 16) return;            // whole wave exits together
    int half = lane >> 4;
    int m = lane & 15;
    int node = wave * 16 + m;
    float ic = 1.0f / fmaxf(cnt[node], 1.0f);

    // A fragment: a[0..7] <- mean2 k = half*8..+7 ; a[8..15] <- h1 k-16 = half*8..+7
    const float4* sp  = (const float4*)(sum2 + node * 16);
    const float4* h1p = (const float4*)(h1 + node * 16);
    float4 s0 = sp[half * 2], s1 = sp[half * 2 + 1];
    float4 g0 = h1p[half * 2], g1 = h1p[half * 2 + 1];
    v16h a;
    a[0] = (_Float16)(s0.x * ic); a[1] = (_Float16)(s0.y * ic);
    a[2] = (_Float16)(s0.z * ic); a[3] = (_Float16)(s0.w * ic);
    a[4] = (_Float16)(s1.x * ic); a[5] = (_Float16)(s1.y * ic);
    a[6] = (_Float16)(s1.z * ic); a[7] = (_Float16)(s1.w * ic);
    a[8]  = (_Float16)g0.x; a[9]  = (_Float16)g0.y;
    a[10] = (_Float16)g0.z; a[11] = (_Float16)g0.w;
    a[12] = (_Float16)g1.x; a[13] = (_Float16)g1.y;
    a[14] = (_Float16)g1.z; a[15] = (_Float16)g1.w;

    int n = lane & 15;
    const v16h* Bf = (const v16h*)W2s;
    for (int nt = 0; nt < 4; ++nt) {
        v16h b = Bf[nt * 32 + lane];
        v8f c = {};
        c = __builtin_amdgcn_wmma_f32_16x16x32_f16(false, a, false, b, (short)0, c, false, false);
        int col = nt * 16 + n;
        float bias = c2_bl[col];
        #pragma unroll
        for (int r = 0; r < 8; ++r) {
            int row = wave * 16 + r + half * 8;
            h2[row * 64 + col] = fmaxf(c[r] + bias, 0.0f);
        }
    }
}

// ---------- x_g partial reduction (mean over nodes of h2) ----------
__global__ void xg_reduce(const float* h2, float* xg) {
    int f = threadIdx.x;                         // 64 threads
    int n0 = blockIdx.x * 256;
    int n1 = n0 + 256; if (n1 > N_NODES) n1 = N_NODES;
    float s = 0.0f;
    for (int n = n0; n < n1; ++n) s += h2[n * 64 + f];
    atomicAdd(&xg[f], s);
}

// ---------- big GEMM via WMMA, fused epilogue:
//   xw = h2(50000x64) @ WihA(64x256) + WihS[:,s] + WihT[:,t] + bih
__global__ __launch_bounds__(256) void xw_gemm_wmma(const float* h2, const _Float16* WihAs,
                                                    const int* e2n_index, const int* e2n,
                                                    const float* WihS, const float* WihT,
                                                    const float* bih, float* xw) {
    int wave = (blockIdx.x * blockDim.x + threadIdx.x) >> 5;
    int lane = threadIdx.x & 31;
    if (wave >= N_NODES / 16) return;
    int half = lane >> 4;
    int m = lane & 15;
    int node = wave * 16 + m;

    // A fragments: each lane's K-octets are contiguous in h2
    const float4* hp = (const float4*)(h2 + node * 64);
    float4 f0 = hp[half * 2],     f1 = hp[half * 2 + 1];      // K  half*8..+7
    float4 f2 = hp[4 + half * 2], f3 = hp[4 + half * 2 + 1];  // K  16+half*8..+7
    float4 f4 = hp[8 + half * 2], f5 = hp[8 + half * 2 + 1];  // K  32+half*8..+7
    float4 f6 = hp[12 + half * 2], f7 = hp[12 + half * 2 + 1];// K  48+half*8..+7
    v16h a0, a1;
    a0[0]=(_Float16)f0.x; a0[1]=(_Float16)f0.y; a0[2]=(_Float16)f0.z; a0[3]=(_Float16)f0.w;
    a0[4]=(_Float16)f1.x; a0[5]=(_Float16)f1.y; a0[6]=(_Float16)f1.z; a0[7]=(_Float16)f1.w;
    a0[8]=(_Float16)f2.x; a0[9]=(_Float16)f2.y; a0[10]=(_Float16)f2.z; a0[11]=(_Float16)f2.w;
    a0[12]=(_Float16)f3.x; a0[13]=(_Float16)f3.y; a0[14]=(_Float16)f3.z; a0[15]=(_Float16)f3.w;
    a1[0]=(_Float16)f4.x; a1[1]=(_Float16)f4.y; a1[2]=(_Float16)f4.z; a1[3]=(_Float16)f4.w;
    a1[4]=(_Float16)f5.x; a1[5]=(_Float16)f5.y; a1[6]=(_Float16)f5.z; a1[7]=(_Float16)f5.w;
    a1[8]=(_Float16)f6.x; a1[9]=(_Float16)f6.y; a1[10]=(_Float16)f6.z; a1[11]=(_Float16)f6.w;
    a1[12]=(_Float16)f7.x; a1[13]=(_Float16)f7.y; a1[14]=(_Float16)f7.z; a1[15]=(_Float16)f7.w;

    // one-hot indices for the 8 rows this lane stores
    int rowbase = wave * 16 + half * 8;
    int sIdx[8], tIdx[8];
    #pragma unroll
    for (int r = 0; r < 8; ++r) {
        int e = e2n_index[rowbase + r];
        sIdx[r] = e2n[e * 2];
        tIdx[r] = e2n[e * 2 + 1];
    }

    int n = lane & 15;
    const v16h* Bf = (const v16h*)WihAs;
    for (int nt = 0; nt < 16; ++nt) {
        v16h b0 = Bf[nt * 32 + lane];            // chunk 0
        v16h b1 = Bf[(16 + nt) * 32 + lane];     // chunk 1
        v8f c = {};
        c = __builtin_amdgcn_wmma_f32_16x16x32_f16(false, a0, false, b0, (short)0, c, false, false);
        c = __builtin_amdgcn_wmma_f32_16x16x32_f16(false, a1, false, b1, (short)0, c, false, false);
        int col = nt * 16 + n;
        float bias = bih[col];
        #pragma unroll
        for (int r = 0; r < 8; ++r) {
            int row = rowbase + r;
            float v = c[r] + bias + WihS[sIdx[r] * 256 + col] + WihT[tIdx[r] * 256 + col];
            xw[row * 256 + col] = v;
        }
    }
}

// ---------- sequential LSTM scan over 50000 steps, one workgroup ----------
__global__ __launch_bounds__(256) void lstm_seq(const float* xw, const float* Whh,
                                                const float* bhh, const float* hx0,
                                                const float* cx0, float* hxcx_out) {
    __shared__ float hxs[64];
    __shared__ float cxs[64];
    __shared__ float gs[256];
    int r = threadIdx.x;
    float w[64];
    #pragma unroll
    for (int j = 0; j < 64; ++j) w[j] = Whh[r * 64 + j];
    float br = bhh[r];
    if (r < 64) { hxs[r] = hx0[r]; cxs[r] = cx0[r]; }
    __syncthreads();
    float xc = xw[r];                             // software pipeline: step 0 input
    for (int i = 0; i < N_NODES; ++i) {
        // issue next step's load before the dependent FMA chain
        float xn = (i + 1 < N_NODES) ? xw[(i + 1) * 256 + r] : 0.0f;
        float g = xc + br;
        #pragma unroll
        for (int j = 0; j < 64; ++j) g = fmaf(w[j], hxs[j], g);
        gs[r] = g;
        __syncthreads();
        if (r < 64) {
            float iv = sigmoidf_(gs[r]);
            float fv = sigmoidf_(gs[64 + r]);
            float gv = tanhf(gs[128 + r]);
            float ov = sigmoidf_(gs[192 + r]);
            float c  = fv * cxs[r] + iv * gv;
            cxs[r] = c;
            hxs[r] = ov * tanhf(c);
        }
        __syncthreads();
        xc = xn;
    }
    if (r < 64) { hxcx_out[r] = hxs[r]; hxcx_out[64 + r] = cxs[r]; }
}

// ---------- head MLP: z=[cx,hx,x_g,onehots] -> 32 -> 16 -> 8 -> 1 ----------
__global__ void head_mlp(const float* hxcx, const float* xg, const int* routing,
                         const float* W0, const float* b0, const float* W1, const float* b1,
                         const float* W2, const float* b2, const float* W3, const float* b3,
                         float* out) {
    __shared__ float z0[32], z1[16], z2[8];
    int t = threadIdx.x;                          // 32 threads
    int s = routing[0], p = routing[1], d = routing[2];
    const float invN = 1.0f / (float)N_NODES;
    {
        const float* w = W0 + t * 384;
        float acc = b0[t];
        for (int j = 0; j < 64; ++j) acc += w[j]       * hxcx[64 + j];   // cx
        for (int j = 0; j < 64; ++j) acc += w[64 + j]  * hxcx[j];        // hx
        for (int j = 0; j < 64; ++j) acc += w[128 + j] * xg[j] * invN;   // x_g
        acc += w[192 + s] + w[256 + p] + w[320 + d];
        z0[t] = fmaxf(acc, 0.0f);
    }
    __syncthreads();
    if (t < 16) {
        float a = b1[t];
        for (int j = 0; j < 32; ++j) a += W1[t * 32 + j] * z0[j];
        z1[t] = fmaxf(a, 0.0f);
    }
    __syncthreads();
    if (t < 8) {
        float a = b2[t];
        for (int j = 0; j < 16; ++j) a += W2[t * 16 + j] * z1[j];
        z2[t] = fmaxf(a, 0.0f);
    }
    __syncthreads();
    if (t == 0) {
        float a = b3[0];
        for (int j = 0; j < 8; ++j) a += W3[j] * z2[j];
        out[0] = fmaxf(a, 0.0f);
    }
}

// ---------- launch ----------
extern "C" void kernel_launch(void* const* d_in, const int* in_sizes, int n_in,
                              void* d_out, int out_size, void* d_ws, size_t ws_size,
                              hipStream_t stream) {
    const float* x        = (const float*)d_in[0];
    const int*   eidx     = (const int*)d_in[1];
    const int*   e2n_idx  = (const int*)d_in[2];
    const int*   e2n      = (const int*)d_in[3];
    const int*   routing  = (const int*)d_in[4];
    const float* hx0      = (const float*)d_in[5];
    const float* cx0      = (const float*)d_in[6];
    const float* c1_Wl    = (const float*)d_in[7];
    const float* c1_bl    = (const float*)d_in[8];
    const float* c1_Wr    = (const float*)d_in[9];
    const float* c2_Wl    = (const float*)d_in[10];
    const float* c2_bl    = (const float*)d_in[11];
    const float* c2_Wr    = (const float*)d_in[12];
    const float* Wih      = (const float*)d_in[13];
    const float* Whh      = (const float*)d_in[14];
    const float* bih      = (const float*)d_in[15];
    const float* bhh      = (const float*)d_in[16];
    const float* W0       = (const float*)d_in[17];
    const float* b0       = (const float*)d_in[18];
    const float* W1       = (const float*)d_in[19];
    const float* b1       = (const float*)d_in[20];
    const float* W2       = (const float*)d_in[21];
    const float* b2       = (const float*)d_in[22];
    const float* W3       = (const float*)d_in[23];
    const float* b3       = (const float*)d_in[24];

    char* ws = (char*)d_ws;
    size_t off = 0;
    auto alloc = [&](size_t bytes) { void* p = ws + off; off = (off + bytes + 255) & ~(size_t)255; return p; };

    // zero-group kept contiguous: sum1 | cnt | sum2 | xg
    float* sum1 = (float*)(ws + off); off += (size_t)N_NODES * 4;
    float* cnt  = (float*)(ws + off); off += (size_t)N_NODES * 4;
    float* sum2 = (float*)(ws + off); off += (size_t)N_NODES * 16 * 4;
    float* xg   = (float*)(ws + off); off += 64 * 4;
    int zero_count = N_NODES + N_NODES + N_NODES * 16 + 64;                  // 900064
    off = (off + 255) & ~(size_t)255;

    float*     h1    = (float*)alloc((size_t)N_NODES * 16 * 4);
    float*     h2    = (float*)alloc((size_t)N_NODES * 64 * 4);
    float*     xw    = (float*)alloc((size_t)N_NODES * 256 * 4);
    float*     hxcx  = (float*)alloc(128 * 4);
    _Float16*  WihAs = (_Float16*)alloc((size_t)2 * 16 * 32 * 16 * 2);       // 32KB swizzled B
    float*     WihS  = (float*)alloc((size_t)64 * 256 * 4);
    float*     WihT  = (float*)alloc((size_t)64 * 256 * 4);
    _Float16*  W2s   = (_Float16*)alloc((size_t)4 * 32 * 16 * 2);            // 4KB swizzled B

    // 1. zero accumulators
    zero_f32<<<(zero_count + 255) / 256, 256, 0, stream>>>(sum1, zero_count);
    // 2. weight prep
    prep_weights<<<(16384 + 255) / 256, 256, 0, stream>>>(Wih, c2_Wl, c2_Wr, WihAs, WihS, WihT, W2s);
    // 3. SAGE1 edges
    sage1_edges<<<(N_EDGES + 255) / 256, 256, 0, stream>>>(eidx, x, sum1, cnt);
    // 4. SAGE1 nodes
    sage1_nodes<<<(N_NODES * 16 + 255) / 256, 256, 0, stream>>>(x, sum1, cnt, c1_Wl, c1_bl, c1_Wr, h1);
    // 5. SAGE2 edges
    {
        long long tot = (long long)N_EDGES * 16;
        sage2_edges<<<(unsigned)((tot + 255) / 256), 256, 0, stream>>>(eidx, h1, sum2);
    }
    // 6. SAGE2 node update (WMMA)
    sage2_wmma<<<(N_NODES / 16 + 7) / 8, 256, 0, stream>>>(sum2, cnt, h1, W2s, c2_bl, h2);
    // 7. x_g reduction
    xg_reduce<<<(N_NODES + 255) / 256, 64, 0, stream>>>(h2, xg);
    // 8. xw GEMM (WMMA) with fused one-hot + bias epilogue
    xw_gemm_wmma<<<(N_NODES / 16 + 7) / 8, 256, 0, stream>>>(h2, WihAs, e2n_idx, e2n,
                                                             WihS, WihT, bih, xw);
    // 9. sequential LSTM
    lstm_seq<<<1, 256, 0, stream>>>(xw, Whh, bhh, hx0, cx0, hxcx);
    // 10. head MLP
    head_mlp<<<1, 32, 0, stream>>>(hxcx, xg, routing, W0, b0, W1, b1, W2, b2, W3, b3, (float*)d_out);

    (void)in_sizes; (void)n_in; (void)out_size; (void)ws_size;
}
